// TmEncoder_69226282877134
// MI455X (gfx1250) — matrix operand order repaired and checked by reading, not compiled, not run on hardware
//
#include <hip/hip_runtime.h>
#include <math.h>

// ---------------------------------------------------------------------------
// TM encoder forward for MI455X (gfx1250, wave32).
// Strategy: the 67MB edge tensor dominates; fuse each layer's edge pipeline
// into one WMMA kernel (bf16 in / f32 accumulate, v_wmma_f32_16x16x32_bf16),
// keeping 16-row tiles in LDS/registers.  ~600MB total HBM traffic -> ~26us.
// Base-pointer + immediate-offset addressing lets row loads/stores clause;
// next-tile global_prefetch_b8 hides HBM latency under WMMA work.
// ---------------------------------------------------------------------------

typedef __attribute__((ext_vector_type(16))) __bf16 v16bf;
typedef __attribute__((ext_vector_type(8)))  float  v8f;

constexpr int NB = 2, NV = 256, NH = 128, TD = 64;
constexpr int ROWS   = NB * NV * NV;   // 131072 edge rows
constexpr int NTILES = ROWS / 16;      // 8192 row tiles
constexpr float LOG1E4 = 9.210340371976184f;
constexpr float TWO_PI = 6.283185307179586f;

__device__ __forceinline__ unsigned short f2bf(float f) {
  union { float f; unsigned u; } c; c.f = f;
  unsigned u = c.u;
  u += 0x7fffu + ((u >> 16) & 1u);          // round-to-nearest-even
  return (unsigned short)(u >> 16);
}

// A-fragment (16x32 bf16, 16-bit A layout, ISA 7.12.2): lane m=L%16,h=L/16,
// elements 0..7 -> K = h*8+e ; elements 8..15 -> K = 16+h*8+e.
__device__ __forceinline__ v16bf lda(const unsigned short* s, int stride, int k0, int kvalid) {
  int lane = threadIdx.x & 31;
  int m = lane & 15, h = lane >> 4;
  union { v16bf v; unsigned short u[16]; } r;
#pragma unroll
  for (int e = 0; e < 8; ++e) {
    int k1 = h * 8 + e;
    int k2 = 16 + h * 8 + e;
    r.u[e]     = (k1 < kvalid) ? s[m * stride + k0 + k1] : (unsigned short)0;
    r.u[8 + e] = (k2 < kvalid) ? s[m * stride + k0 + k2] : (unsigned short)0;
  }
  return r.v;
}

// B-fragment (32x16 bf16): lane n=L%16,h=L/16, elements 0..7 -> K = h*8+e ; elements 8..15 -> K = 16+h*8+e.
// Weight W stored (out,in) row-major: B[k][n] = W[n0+n][k0+k] -> two runs of 8 contiguous per lane.
__device__ __forceinline__ v16bf ldb(const unsigned short* w, int K, int n0, int k0) {
  int lane = threadIdx.x & 31;
  int n = lane & 15, h = lane >> 4;
  const unsigned short* p = w + (size_t)(n0 + n) * K + k0 + h * 8;
  union { v16bf v; unsigned short u[16]; } r;
#pragma unroll
  for (int e = 0; e < 8; ++e) { r.u[e] = p[e]; r.u[8 + e] = p[16 + e]; }
  return r.v;
}

__device__ __forceinline__ v8f wmma_bf(v16bf a, v16bf b, v8f c) {
  return __builtin_amdgcn_wmma_f32_16x16x32_bf16(false, a, false, b, (short)0, c, false, false);
}

// ---------------------------------------------------------------------------
// small helper kernels
// ---------------------------------------------------------------------------
__global__ void k_cvt_bf16(const float* __restrict__ src, unsigned short* __restrict__ dst, int n) {
  for (int i = blockIdx.x * blockDim.x + threadIdx.x; i < n; i += gridDim.x * blockDim.x)
    dst[i] = f2bf(src[i]);
}

__global__ void k_pad_my(const float* __restrict__ src /*16x16*/, unsigned short* __restrict__ dst /*16x32*/) {
  int i = threadIdx.x;
  if (i < 16 * 32) {
    int r = i >> 5, c = i & 31;
    dst[i] = (c < 16) ? f2bf(src[r * 16 + c]) : (unsigned short)0;
  }
}

// h0 = node_embed(pos_embed_sine(v))
__global__ void k_h0(const float* __restrict__ vpos, const float* __restrict__ nw,
                     const float* __restrict__ nb, float* __restrict__ h0) {
  __shared__ float pe[NH];
  int bi = blockIdx.x;       // b*NV + i
  int c  = threadIdx.x;      // 128
  float ye = vpos[bi * 2 + 0] * TWO_PI;
  float xe = vpos[bi * 2 + 1] * TWO_PI;
  float emb = (c < 64) ? ye : xe;
  int cc = c & 63;
  float inv = __expf(-((float)(cc & ~1) / 64.0f) * LOG1E4);
  float ang = emb * inv;
  pe[c] = (c & 1) ? __cosf(ang) : __sinf(ang);
  __syncthreads();
  float acc = nb[c];
  for (int k = 0; k < NH; ++k) acc += pe[k] * nw[c * NH + k];
  h0[bi * NH + c] = acc;
}

// time embedding MLP + per-layer te vectors
__global__ void k_time(const int* __restrict__ ts,
                       const float* __restrict__ te1w, const float* __restrict__ te1b,
                       const float* __restrict__ te2w, const float* __restrict__ te2b,
                       const float* __restrict__ tl0w, const float* __restrict__ tl0b,
                       const float* __restrict__ tl1w, const float* __restrict__ tl1b,
                       float* __restrict__ te /*(2,NB,NH)*/) {
  __shared__ float emb[NH], f1[TD], tf[TD];
  int c = threadIdx.x;       // 128
  for (int b = 0; b < NB; ++b) {
    float t = (float)ts[b];
    {
      int i = c & 63;
      float fr = __expf(-LOG1E4 * (float)i / 64.0f);
      float a = t * fr;
      emb[c] = (c < 64) ? __cosf(a) : __sinf(a);   // concat[cos, sin]
    }
    __syncthreads();
    if (c < TD) {
      float a = te1b[c];
      for (int k = 0; k < NH; ++k) a += emb[k] * te1w[c * NH + k];
      f1[c] = fmaxf(a, 0.f);
    }
    __syncthreads();
    if (c < TD) {
      float a = te2b[c];
      for (int k = 0; k < TD; ++k) a += f1[k] * te2w[c * TD + k];
      tf[c] = fmaxf(a, 0.f);                       // relu(t) used by time_layers
    }
    __syncthreads();
    {
      float a0 = tl0b[c], a1 = tl1b[c];
      for (int k = 0; k < TD; ++k) { float v = tf[k]; a0 += v * tl0w[c * TD + k]; a1 += v * tl1w[c * TD + k]; }
      te[(0 * NB + b) * NH + c] = a0;
      te[(1 * NB + b) * NH + c] = a1;
    }
    __syncthreads();
  }
}

// Q (pre-scaled by 1/sqrt(DF)), K, V projections
__global__ void k_qkv(const float* __restrict__ h,
                      const float* __restrict__ qw, const float* __restrict__ kw,
                      const float* __restrict__ vw,
                      float* __restrict__ Qs, float* __restrict__ Kh, float* __restrict__ Vh) {
  __shared__ float hr[NH];
  int bi = blockIdx.x;
  int o  = threadIdx.x;
  hr[o] = h[bi * NH + o];
  __syncthreads();
  float aq = 0, ak = 0, av = 0;
  for (int k = 0; k < NH; ++k) {
    float v = hr[k];
    aq += v * qw[o * NH + k];
    ak += v * kw[o * NH + k];
    av += v * vw[o * NH + k];
  }
  Qs[bi * NH + o] = aq * 0.25f;
  Kh[bi * NH + o] = ak;
  Vh[bi * NH + o] = av;
}

// ---------------------------------------------------------------------------
// fused sincos embed + edge_embed GEMM (WMMA)
// ---------------------------------------------------------------------------
__global__ void __launch_bounds__(256, 1)
k_edge0(const float* __restrict__ x, const unsigned short* __restrict__ wbf,
        const float* __restrict__ bias, float* __restrict__ eOut) {
  __shared__ unsigned short embS[16 * NH];
  int tid = threadIdx.x;
  int lane = tid & 31, wv = tid >> 5;
  int n = lane & 15, hh = lane >> 4;
  int n0 = wv * 16;
  v16bf wB[4];
#pragma unroll
  for (int kc = 0; kc < 4; ++kc) wB[kc] = ldb(wbf, NH, n0, kc * 32);
  float bb = bias[n0 + n];
  const size_t laneOff = (size_t)(8 * hh) * NH + n0 + n;   // lane's row-half + col

  for (int t = blockIdx.x; t < NTILES; t += gridDim.x) {
    int row0 = t * 16;
    int b = row0 >> 16, rem = row0 & 65535;
    int i = rem >> 8, j0 = rem & 255;
    {
      int e0i = tid * 8;
      int r  = e0i >> 7;
      int c0 = e0i & 127;
      float xv = x[(b * NV + i) * NV + j0 + r];
#pragma unroll
      for (int q = 0; q < 8; ++q) {
        int c = c0 + q;
        float inv = __expf(-((float)(c & ~1) / 128.0f) * LOG1E4);
        float ang = xv * inv;
        embS[r * NH + c] = f2bf((c & 1) ? __cosf(ang) : __sinf(ang));
      }
    }
    // prefetch next tile's x row (gfx1250 global_prefetch_b8 path)
    {
      int tn = t + gridDim.x;
      if (tn < NTILES && tid < 16)
        __builtin_prefetch(&x[(size_t)(tn * 16) + tid], 0, 1);
    }
    __syncthreads();
    v8f acc;
#pragma unroll
    for (int r = 0; r < 8; ++r) acc[r] = bb;
#pragma unroll
    for (int kc = 0; kc < 4; ++kc) acc = wmma_bf(lda(embS, NH, kc * 32, 32), wB[kc], acc);
    // single base address; r*NH folds into the 24-bit instruction offset -> store clause
    float* po = eOut + (size_t)row0 * NH + laneOff;
#pragma unroll
    for (int r = 0; r < 8; ++r) po[r * NH] = acc[r];
    __syncthreads();
  }
}

// ---------------------------------------------------------------------------
// fused per-layer edge pipeline (WMMA): Y, Y1/Y2, newE, (E1/E2, Ym), +te,
// LayerNorm, FFN 128->256->128, residual.
// ---------------------------------------------------------------------------
template <bool LAST>
__global__ void __launch_bounds__(256, 1)
k_edge_layer(const float* __restrict__ eIn, float* __restrict__ eOut,
             float* __restrict__ ymOut,
             const float* __restrict__ Qs, const float* __restrict__ Kh,
             const unsigned short* __restrict__ my1p, const unsigned short* __restrict__ my2p,
             const unsigned short* __restrict__ me1bf, const unsigned short* __restrict__ me2bf,
             const unsigned short* __restrict__ f1wbf, const unsigned short* __restrict__ f2wbf,
             const float* __restrict__ f1b, const float* __restrict__ f2b,
             const float* __restrict__ lng, const float* __restrict__ lnb,
             const float* __restrict__ te /*(NB,NH)*/) {
  __shared__ float sm0f[16 * NH];              // Y f32 -> LN buf -> f1 bf16 (aliased)
  __shared__ unsigned short sm1[16 * NH];      // Y bf16 -> e0 bf16 -> eln bf16
  __shared__ float sMean[16], sRstd[16];
  unsigned short* f1S = (unsigned short*)sm0f; // 16 x 256 bf16

  int tid = threadIdx.x;
  int lane = tid & 31, wv = tid >> 5;          // wave == head == col-tile
  int n = lane & 15, hh = lane >> 4;
  int n0 = wv * 16;
  const size_t laneOff = (size_t)(8 * hh) * NH + n0 + n;

  v16bf bMy1 = ldb(my1p, 32, 0, 0);
  v16bf bMy2 = ldb(my2p, 32, 0, 0);
  v16bf bMe1[4], bMe2[4];
  if constexpr (!LAST) {
#pragma unroll
    for (int kc = 0; kc < 4; ++kc) { bMe1[kc] = ldb(me1bf, NH, n0, kc * 32); bMe2[kc] = ldb(me2bf, NH, n0, kc * 32); }
  }
  v16bf bF1[2][4];
#pragma unroll
  for (int ct = 0; ct < 2; ++ct)
#pragma unroll
    for (int kc = 0; kc < 4; ++kc) bF1[ct][kc] = ldb(f1wbf, NH, ct * NH + n0, kc * 32);
  v16bf bF2[8];
#pragma unroll
  for (int kc = 0; kc < 8; ++kc) bF2[kc] = ldb(f2wbf, 2 * NH, n0, kc * 32);

  float bb1[2] = { f1b[n0 + n], f1b[NH + n0 + n] };
  float bb2 = f2b[n0 + n];

  for (int t = blockIdx.x; t < NTILES; t += gridDim.x) {
    int row0 = t * 16;
    int b = row0 >> 16, rem = row0 & 65535;
    int i = rem >> 8, j0 = rem & 255;
    float tec = te[b * NH + n0 + n];

    // prefetch next tile's e_in rows (gfx1250 global_prefetch_b8, no counter)
    {
      int tn = t + gridDim.x;
      if (tn < NTILES) {
        const float* pp = eIn + (size_t)(tn * 16) * NH + laneOff;
#pragma unroll
        for (int r = 0; r < 8; ++r) __builtin_prefetch(&pp[r * NH], 0, 1);
      }
    }

    // ---- Y tile: Y[j][c] = Qs[b,i,c] * Kh[b,j,c]  (f32 + bf16 copies)
    {
      int r  = tid >> 4;
      int c0 = (tid & 15) * 8;
      const float* qrow = Qs + (size_t)(b * NV + i) * NH;
      const float* krow = Kh + (size_t)(b * NV + j0 + r) * NH;
#pragma unroll
      for (int q = 0; q < 8; ++q) {
        int c = c0 + q;
        float y = qrow[c] * krow[c];
        sm0f[r * NH + c] = y;
        sm1[r * NH + c]  = f2bf(y);
      }
    }
    __syncthreads();

    // ---- Y1/Y2: per-head 16x16 GEMM (K padded 16->32)
    v8f y1, y2;
#pragma unroll
    for (int r = 0; r < 8; ++r) { y1[r] = 0.f; y2[r] = 0.f; }
    {
      v16bf aY = lda(sm1, NH, wv * 16, 16);
      y1 = wmma_bf(aY, bMy1, y1);
      y2 = wmma_bf(aY, bMy2, y2);
    }

    // ---- load e0 (base + immediate offsets -> load clause), newE = e0*(Y1+1)+Y2
    v8f e0, nE;
    {
      const float* pi = eIn + (size_t)row0 * NH + laneOff;
#pragma unroll
      for (int r = 0; r < 8; ++r) e0[r] = pi[r * NH];
    }
#pragma unroll
    for (int r = 0; r < 8; ++r) nE[r] = e0[r] * (y1[r] + 1.f) + y2[r];
    __syncthreads();                           // done reading Y bf16

    // ---- e0 -> bf16 LDS (A for me1/me2)
#pragma unroll
    for (int r = 0; r < 8; ++r) sm1[(r + 8 * hh) * NH + n0 + n] = f2bf(e0[r]);
    __syncthreads();

    if constexpr (!LAST) {
      v8f E1, E2;
#pragma unroll
      for (int r = 0; r < 8; ++r) { E1[r] = 0.f; E2[r] = 0.f; }
#pragma unroll
      for (int kc = 0; kc < 4; ++kc) {
        v16bf a = lda(sm1, NH, kc * 32, 32);
        E1 = wmma_bf(a, bMe1[kc], E1);
        E2 = wmma_bf(a, bMe2[kc], E2);
      }
      float* py = ymOut + (size_t)row0 * NH + laneOff;
#pragma unroll
      for (int r = 0; r < 8; ++r) {
        float yv = sm0f[(r + 8 * hh) * NH + n0 + n];
        py[r * NH] = yv * (E1[r] + 1.f) + E2[r];
      }
    }
    __syncthreads();

    // ---- LayerNorm input: newE + te
#pragma unroll
    for (int r = 0; r < 8; ++r) sm0f[(r + 8 * hh) * NH + n0 + n] = nE[r] + tec;
    __syncthreads();
    if (tid < 16) {
      float s = 0.f, q = 0.f;
      for (int k = 0; k < NH; ++k) { float v = sm0f[tid * NH + k]; s += v; q += v * v; }
      float m = s * (1.f / NH);
      float var = q * (1.f / NH) - m * m;
      sMean[tid] = m;
      sRstd[tid] = rsqrtf(fmaxf(var, 0.f) + 1e-5f);
    }
    __syncthreads();
    {
      int r  = tid >> 4;
      int c0 = (tid & 15) * 8;
      float m = sMean[r], rs = sRstd[r];
#pragma unroll
      for (int q = 0; q < 8; ++q) {
        int c = c0 + q;
        float v = (sm0f[r * NH + c] - m) * rs * lng[c] + lnb[c];
        sm1[r * NH + c] = f2bf(v);
      }
    }
    __syncthreads();

    // ---- FFN1 (128 -> 256) + relu, bf16 into f1S (aliases sm0f)
#pragma unroll
    for (int ct = 0; ct < 2; ++ct) {
      v8f f;
#pragma unroll
      for (int r = 0; r < 8; ++r) f[r] = bb1[ct];
#pragma unroll
      for (int kc = 0; kc < 4; ++kc) f = wmma_bf(lda(sm1, NH, kc * 32, 32), bF1[ct][kc], f);
#pragma unroll
      for (int r = 0; r < 8; ++r)
        f1S[(r + 8 * hh) * (2 * NH) + ct * NH + n0 + n] = f2bf(fmaxf(f[r], 0.f));
    }
    __syncthreads();

    // ---- FFN2 (256 -> 128) + residual
    v8f f2;
#pragma unroll
    for (int r = 0; r < 8; ++r) f2[r] = bb2;
#pragma unroll
    for (int kc = 0; kc < 8; ++kc) f2 = wmma_bf(lda(f1S, 2 * NH, kc * 32, 32), bF2[kc], f2);
    {
      float* po = eOut + (size_t)row0 * NH + laneOff;
#pragma unroll
      for (int r = 0; r < 8; ++r) po[r * NH] = e0[r] + f2[r];
    }
    __syncthreads();
  }
}

// softmax over keys (j) + weighted sum with Vh; second Ym read hits L2 (67MB < 192MB)
__global__ void k_softmax(const float* __restrict__ ym, const float* __restrict__ Vh,
                          float* __restrict__ hOut) {
  int bi = blockIdx.x;           // b*NV + i
  int b = bi >> 8;
  int c = threadIdx.x;           // 128
  const float* base = ym + (size_t)bi * NV * NH + c;
  float m = -3.4e38f, s = 0.f;
  for (int j = 0; j < NV; ++j) {
    float v = base[(size_t)j * NH];
    float nm = fmaxf(m, v);
    s = s * __expf(m - nm) + __expf(v - nm);
    m = nm;
  }
  float inv = 1.f / s;
  const float* vbase = Vh + (size_t)(b * NV) * NH + c;
  float acc = 0.f;
  for (int j = 0; j < NV; ++j)
    acc += __expf(base[(size_t)j * NH] - m) * inv * vbase[(size_t)j * NH];
  hOut[(size_t)bi * NH + c] = acc;
}

// h1 = h0 + ffn_h(LN(h_attn))
__global__ void k_hffn(const float* __restrict__ h0, const float* __restrict__ hAttn,
                       const float* __restrict__ lng, const float* __restrict__ lnb,
                       const float* __restrict__ f1w, const float* __restrict__ f1b,
                       const float* __restrict__ f2w, const float* __restrict__ f2b,
                       float* __restrict__ h1) {
  __shared__ float raw[NH], ln[NH], f1[2 * NH];
  __shared__ float sM, sR;
  int bi = blockIdx.x;
  int tid = threadIdx.x;         // 256
  if (tid < NH) raw[tid] = hAttn[bi * NH + tid];
  __syncthreads();
  if (tid == 0) {
    float s = 0, q = 0;
    for (int k = 0; k < NH; ++k) { float v = raw[k]; s += v; q += v * v; }
    float m = s / NH;
    sM = m;
    sR = rsqrtf(fmaxf(q / NH - m * m, 0.f) + 1e-5f);
  }
  __syncthreads();
  if (tid < NH) ln[tid] = (raw[tid] - sM) * sR * lng[tid] + lnb[tid];
  __syncthreads();
  {
    float a = f1b[tid];
    for (int k = 0; k < NH; ++k) a += ln[k] * f1w[tid * NH + k];
    f1[tid] = fmaxf(a, 0.f);
  }
  __syncthreads();
  if (tid < NH) {
    float a = f2b[tid];
    for (int k = 0; k < 2 * NH; ++k) a += f1[k] * f2w[tid * 2 * NH + k];
    h1[bi * NH + tid] = h0[bi * NH + tid] + a;
  }
}

// GroupNorm reduction (3 stages) + fused relu + 1x1 conv output
__global__ void k_zero(float* p, int n) {
  int i = blockIdx.x * blockDim.x + threadIdx.x;
  if (i < n) p[i] = 0.f;
}

__global__ void k_gnpart(const float* __restrict__ e, float* __restrict__ stats) {
  __shared__ float ls[64], lq[64];
  int tid = threadIdx.x;
  if (tid < 64) { ls[tid] = 0.f; lq[tid] = 0.f; }
  __syncthreads();
  size_t base = (size_t)blockIdx.x * 8192;
  for (int t = tid; t < 8192; t += 256) {
    size_t idx = base + t;
    float v = e[idx];
    int c = (int)(idx & 127);
    int b = (int)(idx >> 23);
    int key = b * 32 + (c >> 2);
    atomicAdd(&ls[key], v);
    atomicAdd(&lq[key], v * v);
  }
  __syncthreads();
  if (tid < 64) { atomicAdd(&stats[tid], ls[tid]); atomicAdd(&stats[64 + tid], lq[tid]); }
}

__global__ void k_gnfin(const float* __restrict__ stats, float* __restrict__ msr) {
  int g = threadIdx.x;
  if (g < 64) {
    const float invN = 1.f / (4.f * NV * NV);
    float m = stats[g] * invN;
    float var = stats[64 + g] * invN - m * m;
    msr[g * 2]     = m;
    msr[g * 2 + 1] = rsqrtf(fmaxf(var, 0.f) + 1e-5f);
  }
}

__global__ void k_out(const float* __restrict__ e, const float* __restrict__ msr,
                      const float* __restrict__ gng, const float* __restrict__ gnb,
                      const float* __restrict__ cw, const float* __restrict__ cb,
                      float* __restrict__ out) {
  __shared__ float ca[NH], cbv[NH], cwv[NH];
  int bi = blockIdx.x;            // b*NV + i
  int b = bi >> 8;
  int tid = threadIdx.x;          // 256
  if (tid < NH) {
    int g = b * 32 + (tid >> 2);
    float m = msr[g * 2], r = msr[g * 2 + 1];
    float a = r * gng[tid];
    ca[tid]  = a;
    cbv[tid] = gnb[tid] - m * a;
    cwv[tid] = cw[tid];
  }
  __syncthreads();
  int j = tid;
  const float* row = e + ((size_t)bi * NV + j) * NH;
  float acc = cb[0];
  for (int c = 0; c < NH; ++c) {
    float v = row[c] * ca[c] + cbv[c];
    acc += fmaxf(v, 0.f) * cwv[c];
  }
  out[(size_t)bi * NV + j] = acc;
}

// ---------------------------------------------------------------------------
extern "C" void kernel_launch(void* const* d_in, const int* in_sizes, int n_in,
                              void* d_out, int out_size, void* d_ws, size_t ws_size,
                              hipStream_t stream) {
  (void)in_sizes; (void)n_in; (void)out_size; (void)ws_size;
  // Input order: jax pytree flattening (sorted dict keys; capitals before lowercase).
  const float* x       = (const float*)d_in[0];
  const int*   ts      = (const int*)d_in[1];
  const float* vpos    = (const float*)d_in[2];
  const float* conv_b  = (const float*)d_in[3];
  const float* conv_w  = (const float*)d_in[4];
  const float* edge_b  = (const float*)d_in[5];
  const float* edge_w  = (const float*)d_in[6];
  const float* gn_b    = (const float*)d_in[7];
  const float* gn_g    = (const float*)d_in[8];
  const float* l0_Kw   = (const float*)d_in[9];
  const float* l0_Qw   = (const float*)d_in[10];
  const float* l0_Vw   = (const float*)d_in[11];
  const float* l0_me1  = (const float*)d_in[12];
  const float* l0_me2  = (const float*)d_in[13];
  const float* l0_my1  = (const float*)d_in[14];
  const float* l0_my2  = (const float*)d_in[15];
  const float* l0_f1b  = (const float*)d_in[16];
  const float* l0_f1w  = (const float*)d_in[17];
  const float* l0_f2b  = (const float*)d_in[18];
  const float* l0_f2w  = (const float*)d_in[19];
  const float* l0_h1b  = (const float*)d_in[20];
  const float* l0_h1w  = (const float*)d_in[21];
  const float* l0_h2b  = (const float*)d_in[22];
  const float* l0_h2w  = (const float*)d_in[23];
  const float* l0_lneb = (const float*)d_in[24];
  const float* l0_lneg = (const float*)d_in[25];
  const float* l0_lnhb = (const float*)d_in[26];
  const float* l0_lnhg = (const float*)d_in[27];
  const float* l1_Kw   = (const float*)d_in[28];
  const float* l1_Qw   = (const float*)d_in[29];
  const float* l1_Vw   = (const float*)d_in[30];
  const float* l1_my1  = (const float*)d_in[31];
  const float* l1_my2  = (const float*)d_in[32];
  const float* l1_f1b  = (const float*)d_in[33];
  const float* l1_f1w  = (const float*)d_in[34];
  const float* l1_f2b  = (const float*)d_in[35];
  const float* l1_f2w  = (const float*)d_in[36];
  const float* l1_lneb = (const float*)d_in[37];
  const float* l1_lneg = (const float*)d_in[38];
  const float* node_b  = (const float*)d_in[39];
  const float* node_w  = (const float*)d_in[40];
  const float* te1b    = (const float*)d_in[41];
  const float* te1w    = (const float*)d_in[42];
  const float* te2b    = (const float*)d_in[43];
  const float* te2w    = (const float*)d_in[44];
  const float* tl0b    = (const float*)d_in[45];
  const float* tl0w    = (const float*)d_in[46];
  const float* tl1b    = (const float*)d_in[47];
  const float* tl1w    = (const float*)d_in[48];

  char* wsb = (char*)d_ws;
  size_t off = 0;
  auto alloc = [&](size_t bytes) -> void* {
    void* p = wsb + off;
    off += (bytes + 255) & ~(size_t)255;
    return p;
  };
  const size_t EBYTES = (size_t)ROWS * NH * sizeof(float);   // 67MB
  float* eA    = (float*)alloc(EBYTES);
  float* eB    = (float*)alloc(EBYTES);
  float* ymB   = (float*)alloc(EBYTES);
  float* h0    = (float*)alloc((size_t)NB * NV * NH * 4);
  float* h1    = (float*)alloc((size_t)NB * NV * NH * 4);
  float* hAtt  = (float*)alloc((size_t)NB * NV * NH * 4);
  float* Qs0   = (float*)alloc((size_t)NB * NV * NH * 4);
  float* Kh0   = (float*)alloc((size_t)NB * NV * NH * 4);
  float* Vh0   = (float*)alloc((size_t)NB * NV * NH * 4);
  float* Qs1   = (float*)alloc((size_t)NB * NV * NH * 4);
  float* Kh1   = (float*)alloc((size_t)NB * NV * NH * 4);
  float* Vh1   = (float*)alloc((size_t)NB * NV * NH * 4);
  float* teBuf = (float*)alloc(2 * NB * NH * 4);
  float* gnSt  = (float*)alloc(128 * 4);
  float* gnMsr = (float*)alloc(128 * 4);
  unsigned short* edgeWbf = (unsigned short*)alloc(16384 * 2);
  unsigned short* me1bf   = (unsigned short*)alloc(16384 * 2);
  unsigned short* me2bf   = (unsigned short*)alloc(16384 * 2);
  unsigned short* my1p0   = (unsigned short*)alloc(512 * 2);
  unsigned short* my2p0   = (unsigned short*)alloc(512 * 2);
  unsigned short* my1p1   = (unsigned short*)alloc(512 * 2);
  unsigned short* my2p1   = (unsigned short*)alloc(512 * 2);
  unsigned short* f1w0bf  = (unsigned short*)alloc(32768 * 2);
  unsigned short* f2w0bf  = (unsigned short*)alloc(32768 * 2);
  unsigned short* f1w1bf  = (unsigned short*)alloc(32768 * 2);
  unsigned short* f2w1bf  = (unsigned short*)alloc(32768 * 2);

  // weight conversions to bf16
  k_cvt_bf16<<<64, 256, 0, stream>>>(edge_w, edgeWbf, 16384);
  k_cvt_bf16<<<64, 256, 0, stream>>>(l0_me1, me1bf, 16384);
  k_cvt_bf16<<<64, 256, 0, stream>>>(l0_me2, me2bf, 16384);
  k_cvt_bf16<<<64, 256, 0, stream>>>(l0_f1w, f1w0bf, 32768);
  k_cvt_bf16<<<64, 256, 0, stream>>>(l0_f2w, f2w0bf, 32768);
  k_cvt_bf16<<<64, 256, 0, stream>>>(l1_f1w, f1w1bf, 32768);
  k_cvt_bf16<<<64, 256, 0, stream>>>(l1_f2w, f2w1bf, 32768);
  k_pad_my<<<1, 512, 0, stream>>>(l0_my1, my1p0);
  k_pad_my<<<1, 512, 0, stream>>>(l0_my2, my2p0);
  k_pad_my<<<1, 512, 0, stream>>>(l1_my1, my1p1);
  k_pad_my<<<1, 512, 0, stream>>>(l1_my2, my2p1);

  // small front-end
  k_h0<<<NB * NV, NH, 0, stream>>>(vpos, node_w, node_b, h0);
  k_time<<<1, NH, 0, stream>>>(ts, te1w, te1b, te2w, te2b, tl0w, tl0b, tl1w, tl1b, teBuf);
  k_qkv<<<NB * NV, NH, 0, stream>>>(h0, l0_Qw, l0_Kw, l0_Vw, Qs0, Kh0, Vh0);

  // edge tensor pipeline
  k_edge0<<<1024, 256, 0, stream>>>(x, edgeWbf, edge_b, eA);
  k_edge_layer<false><<<1024, 256, 0, stream>>>(
      eA, eB, ymB, Qs0, Kh0, my1p0, my2p0, me1bf, me2bf,
      f1w0bf, f2w0bf, l0_f1b, l0_f2b, l0_lneg, l0_lneb, teBuf + 0 * NB * NH);
  k_softmax<<<NB * NV, NH, 0, stream>>>(ymB, Vh0, hAtt);
  k_hffn<<<NB * NV, 256, 0, stream>>>(h0, hAtt, l0_lnhg, l0_lnhb,
                                      l0_h1w, l0_h1b, l0_h2w, l0_h2b, h1);
  k_qkv<<<NB * NV, NH, 0, stream>>>(h1, l1_Qw, l1_Kw, l1_Vw, Qs1, Kh1, Vh1);
  k_edge_layer<true><<<1024, 256, 0, stream>>>(
      eB, eA, nullptr, Qs1, Kh1, my1p1, my2p1, nullptr, nullptr,
      f1w1bf, f2w1bf, l1_f1b, l1_f2b, l1_lneg, l1_lneb, teBuf + 1 * NB * NH);

  // GroupNorm + relu + 1x1 conv
  k_zero<<<1, 128, 0, stream>>>(gnSt, 128);
  k_gnpart<<<2048, 256, 0, stream>>>(eA, gnSt);
  k_gnfin<<<1, 64, 0, stream>>>(gnSt, gnMsr);
  k_out<<<NB * NV, 256, 0, stream>>>(eA, gnMsr, gn_g, gn_b, conv_w, conv_b, (float*)d_out);
}
